// BasicTransformerBlockST_2628519985165
// MI455X (gfx1250) — matrix operand order repaired and checked by reading, not compiled, not run on hardware
//
#include <hip/hip_runtime.h>
#include <math.h>

typedef __attribute__((ext_vector_type(16))) _Float16 v16h;
typedef __attribute__((ext_vector_type(8)))  float    v8f;
typedef __attribute__((ext_vector_type(4)))  unsigned int u32x4;
typedef __attribute__((ext_vector_type(8)))  int      i32x8;
typedef __attribute__((ext_vector_type(4)))  int      i32x4;

#define R_TOK   18432      // total tokens (b*t*h*w)
#define CDIM    640
#define NHEADS  8
#define DHEAD   80
#define TT      16
#define HWN     576
#define NB      2
#define BTN     32
#define BHW     1152
#define CTXD    1024
#define CTXN    77
#define CTXP    80
#define FF1N    5120
#define FFIN    2560

#if __has_builtin(__builtin_amdgcn_tensor_load_to_lds) && __has_builtin(__builtin_amdgcn_s_wait_tensorcnt)
#define USE_TDM 1
#endif

union H16 { uint4 u[2]; v16h h; };
union H8  { uint4 u; _Float16 h[8]; };

static __device__ __forceinline__ v8f v8f_zero() {
  v8f z; for (int i = 0; i < 8; ++i) z[i] = 0.f; return z;
}

// ---------------------------------------------------------------- elementwise
__global__ void cvt16_kernel(const float* __restrict__ src, _Float16* __restrict__ dst, long long n) {
  long long i = (long long)blockIdx.x * blockDim.x + threadIdx.x;
  if (i < n) dst[i] = (_Float16)src[i];
}

// w[K][N] (row major) -> wT[N][K]
__global__ void cvtT16_kernel(const float* __restrict__ src, _Float16* __restrict__ dst,
                              int K, int N) {
  long long i = (long long)blockIdx.x * blockDim.x + threadIdx.x;
  long long n = (long long)K * N;
  if (i >= n) return;
  int nn = (int)(i / K);
  int kk = (int)(i % K);
  dst[i] = (_Float16)src[(size_t)kk * N + nn];
}

__global__ void cvt_ctx_kernel(const float* __restrict__ ctx, _Float16* __restrict__ dst) {
  long long i = (long long)blockIdx.x * blockDim.x + threadIdx.x;
  long long n = (long long)NB * CTXP * CTXD;
  if (i >= n) return;
  int c = (int)(i % CTXD);
  int r = (int)((i / CTXD) % CTXP);
  int b = (int)(i / ((long long)CTXD * CTXP));
  dst[i] = (r < CTXN) ? (_Float16)ctx[((size_t)b * CTXN + r) * CTXD + c] : (_Float16)0.f;
}

// x [b,c,t,h,w] -> xs [(b t),(h w),c]
__global__ void permute_in_kernel(const float* __restrict__ x, float* __restrict__ xs) {
  long long i = (long long)blockIdx.x * blockDim.x + threadIdx.x;
  long long n = (long long)R_TOK * CDIM;
  if (i >= n) return;
  int hw = (int)(i % HWN);
  long long r = i / HWN;
  int t = (int)(r % TT); r /= TT;
  int c = (int)(r % CDIM);
  int b = (int)(r / CDIM);
  xs[(((size_t)(b * TT + t)) * HWN + hw) * CDIM + c] = x[i];
}

__global__ void permute_out_kernel(const float* __restrict__ xs, float* __restrict__ out) {
  long long i = (long long)blockIdx.x * blockDim.x + threadIdx.x;
  long long n = (long long)R_TOK * CDIM;
  if (i >= n) return;
  int hw = (int)(i % HWN);
  long long r = i / HWN;
  int t = (int)(r % TT); r /= TT;
  int c = (int)(r % CDIM);
  int b = (int)(r / CDIM);
  out[i] = xs[(((size_t)(b * TT + t)) * HWN + hw) * CDIM + c];
}

// dir 0: spatial -> temporal ; dir 1: temporal -> spatial
__global__ void s2t_kernel(const float* __restrict__ in, float* __restrict__ out, int dir) {
  long long i = (long long)blockIdx.x * blockDim.x + threadIdx.x;
  long long n = (long long)R_TOK * CDIM;
  if (i >= n) return;
  int c = (int)(i % CDIM);
  long long r = i / CDIM;
  int b = (int)(r / (TT * HWN));
  int q = (int)(r % (TT * HWN));
  int t = q / HWN;
  int hw = q % HWN;
  size_t rt = ((size_t)b * HWN + hw) * TT + t;
  if (dir == 0) out[rt * CDIM + c] = in[i];
  else          out[i] = in[rt * CDIM + c];
}

__global__ void geglu_kernel(const _Float16* __restrict__ p, _Float16* __restrict__ g) {
  long long i = (long long)blockIdx.x * blockDim.x + threadIdx.x;
  long long n = (long long)R_TOK * FFIN;
  if (i >= n) return;
  long long row = i / FFIN;
  int j = (int)(i % FFIN);
  float a  = (float)p[row * FF1N + j];
  float x  = (float)p[row * FF1N + FFIN + j];
  float ge = 0.5f * x * (1.0f + erff(x * 0.70710678118654752f));
  g[i] = (_Float16)(a * ge);
}

// ---------------------------------------------------------------- layernorm
__global__ __launch_bounds__(128)
void ln_rows_kernel(const float* __restrict__ X, const float* __restrict__ g,
                    const float* __restrict__ b, _Float16* __restrict__ Y) {
  int row = blockIdx.x;
  int tid = threadIdx.x;
  const float* xr = X + (size_t)row * CDIM;
  float v[5];
  float s = 0.f;
  for (int i = 0; i < 5; ++i) { v[i] = xr[tid + 128 * i]; s += v[i]; }
  __shared__ float red[128];
  red[tid] = s; __syncthreads();
  for (int o = 64; o > 0; o >>= 1) { if (tid < o) red[tid] += red[tid + o]; __syncthreads(); }
  float mean = red[0] / (float)CDIM;
  __syncthreads();
  float qs = 0.f;
  for (int i = 0; i < 5; ++i) { float d = v[i] - mean; qs += d * d; }
  red[tid] = qs; __syncthreads();
  for (int o = 64; o > 0; o >>= 1) { if (tid < o) red[tid] += red[tid + o]; __syncthreads(); }
  float rstd = rsqrtf(red[0] / (float)CDIM + 1e-5f);
  for (int i = 0; i < 5; ++i) {
    int c = tid + 128 * i;
    Y[(size_t)row * CDIM + c] = (_Float16)((v[i] - mean) * rstd * g[c] + b[c]);
  }
}

// ---------------------------------------------------------------- WMMA GEMM
// C[M,N] = A[M,K](f16,row-major) * W[K,N] given as WT[N][K] (f16).
// Epilogue: +bias, +res, write f32 (Cf) / f16 (Ch) / f16-transposed (ChT, stride ldct).
#define GBM 128
#define GBN 64
#define GBK 32
#define BSTR 40   // halves; 80B row stride -> 16B aligned frag reads

#ifdef USE_TDM
// Issue a TDM 2D tile load: 64 rows x 32 halves from WT (row stride K halves)
// into LDS at lds_off, padded +16B per 64B row => BSTR stride. Per ISA 08 §8.
static __device__ __forceinline__ void tdm_load_tile(const _Float16* gsrc, unsigned lds_off,
                                                     int d0rem, int d1rem, int K) {
  unsigned long long ga = (unsigned long long)(size_t)gsrc;
  u32x4 g0;
  g0[0] = 1u;                                    // count=1, user descriptor
  g0[1] = lds_off;                               // lds_addr [63:32]
  g0[2] = (unsigned)(ga & 0xFFFFFFFFull);        // global_addr [95:64]
  g0[3] = (unsigned)((ga >> 32) & 0x01FFFFFFull) | (2u << 30);  // addr[56:32], type=2
  i32x8 g1;
  g1[0] = (int)((1u << 16) | (1u << 20) | (3u << 22) | (3u << 25)); // 2B elems, pad 4DW per 16DW
  g1[1] = (int)(((unsigned)d0rem & 0xFFFFu) << 16);                 // tensor_dim0 lo
  g1[2] = (int)((((unsigned)d0rem >> 16) & 0xFFFFu) | (((unsigned)d1rem & 0xFFFFu) << 16));
  g1[3] = (int)((((unsigned)d1rem >> 16) & 0xFFFFu) | (32u << 16)); // tile_dim0 = 32
  g1[4] = 64;                                                       // tile_dim1 = 64
  g1[5] = K;                                                        // tensor_dim0_stride lo
  g1[6] = 0;
  g1[7] = 0;
  i32x4 gz; gz[0] = 0; gz[1] = 0; gz[2] = 0; gz[3] = 0;
  i32x8 gz8;
  for (int i = 0; i < 8; ++i) gz8[i] = 0;
  // 6-arg form on this toolchain (clang-23 / therock-10.0 headers)
  __builtin_amdgcn_tensor_load_to_lds(g0, g1, gz, gz, gz8, 0);
}
#endif

__global__ __launch_bounds__(256)
void gemm_f16_kernel(const _Float16* __restrict__ A, const _Float16* __restrict__ BwT,
                     const float* __restrict__ bias, const float* __restrict__ res,
                     float* __restrict__ Cf, _Float16* __restrict__ Ch,
                     _Float16* __restrict__ ChT, int M, int N, int K, int ldct) {
  __shared__ _Float16 Bt[2][GBN * BSTR];
  int tid  = threadIdx.x;
  int lane = tid & 31;
  int wave = tid >> 5;
  int l15  = lane & 15;
  int base = (lane < 16) ? 0 : 8;
  int bm = blockIdx.x * GBM;
  int bn = blockIdx.y * GBN;

  int grow = bm + 16 * wave + l15;
  if (grow >= M) grow = M - 1;     // stay inside buffer; bad rows masked at store
  const _Float16* arow = A + (size_t)grow * K;

  v8f acc[4];
  for (int t = 0; t < 4; ++t) acc[t] = v8f_zero();

#ifdef USE_TDM
  unsigned lds0 = (unsigned)(size_t)(void*)&Bt[0][0];
  unsigned lds1 = (unsigned)(size_t)(void*)&Bt[1][0];
  const _Float16* wtile = BwT + (size_t)bn * K;
  int nsteps = K / GBK;
  if (wave == 0) tdm_load_tile(wtile, lds0, K, N - bn, K);
  for (int s = 0; s < nsteps; ++s) {
    int k0 = s * GBK;
    if (wave == 0) __builtin_amdgcn_s_wait_tensorcnt(0);
    __syncthreads();                       // tile s visible to all waves
    if (wave == 0 && s + 1 < nsteps)
      tdm_load_tile(wtile + k0 + GBK, (s & 1) ? lds0 : lds1, K - k0 - GBK, N - bn, K);
    __builtin_prefetch(arow + k0 + GBK, 0, 1);
    const _Float16* Bbuf = Bt[s & 1];
    H16 af;
    af.u[0] = *(const uint4*)(arow + k0 + 2 * base);
    af.u[1] = *(const uint4*)(arow + k0 + 2 * base + 8);
    for (int t = 0; t < 4; ++t) {
      const _Float16* bp = Bbuf + (t * 16 + l15) * BSTR + ((lane < 16) ? 0 : 16);
      H16 bf;
      bf.u[0] = *(const uint4*)(bp);
      bf.u[1] = *(const uint4*)(bp + 8);
      acc[t] = __builtin_amdgcn_wmma_f32_16x16x32_f16(false, af.h, false, bf.h,
                                                      (short)0, acc[t], false, false);
    }
    __syncthreads();                       // reads done before buffer reuse
  }
#else
  for (int k0 = 0; k0 < K; k0 += GBK) {
    __syncthreads();
    // stage Wt tile: 64 rows(n) x 32 halves(k), coalesced along k
    for (int i = 0; i < 2; ++i) {
      int c = tid + 256 * i;
      int n = c >> 3;
      int o = (c & 7) * 4;
      *(uint2*)(&Bt[0][n * BSTR + o]) = *(const uint2*)(BwT + (size_t)(bn + n) * K + k0 + o);
    }
    __builtin_prefetch(arow + k0 + GBK, 0, 1);
    __builtin_prefetch(BwT + (size_t)(bn + (tid >> 2)) * K + k0 + GBK, 0, 1);
    __syncthreads();
    H16 af;
    af.u[0] = *(const uint4*)(arow + k0 + 2 * base);
    af.u[1] = *(const uint4*)(arow + k0 + 2 * base + 8);
    for (int t = 0; t < 4; ++t) {
      const _Float16* bp = &Bt[0][(t * 16 + l15) * BSTR + ((lane < 16) ? 0 : 16)];
      H16 bf;
      bf.u[0] = *(const uint4*)(bp);
      bf.u[1] = *(const uint4*)(bp + 8);
      acc[t] = __builtin_amdgcn_wmma_f32_16x16x32_f16(false, af.h, false, bf.h,
                                                      (short)0, acc[t], false, false);
    }
  }
#endif
  for (int t = 0; t < 4; ++t) {
    int col = bn + t * 16 + l15;
    for (int r = 0; r < 8; ++r) {
      int row = bm + 16 * wave + ((lane < 16) ? r : r + 8);
      if (row < M) {
        float v = acc[t][r];
        if (bias) v += bias[col];
        size_t idx = (size_t)row * N + col;
        if (res) v += res[idx];
        if (ChT)      ChT[(size_t)col * ldct + row] = (_Float16)v;
        else if (Cf)  Cf[idx] = v;
        else          Ch[idx] = (_Float16)v;
      }
    }
  }
}

// ---------------------------------------------------------------- flash attention
// One wave per (batch, head, 16-query tile). S^T = K Q^T ; O^T = V^T P^T.
// No LDS: every fragment is two 16B global loads (D=80 is a multiple of 16).
// V supplied pre-transposed: Vt[channel][token], stride ldv tokens.
__global__ __launch_bounds__(32)
void flash_attn_kernel(const _Float16* __restrict__ Qm, const _Float16* __restrict__ Km,
                       const _Float16* __restrict__ Vt, _Float16* __restrict__ Om,
                       int Nq, int Nkv, int kv_div, int kv_stride, int ldv, float scale) {
  int qtiles = Nq >> 4;
  int bid   = blockIdx.x;
  int qt    = bid % qtiles;
  int head  = (bid / qtiles) % NHEADS;
  int batch = bid / (qtiles * NHEADS);
  int lane = threadIdx.x;
  int l15  = lane & 15;
  int hi   = lane >> 4;
  int base = hi ? 8 : 0;
  const uint4 z4 = make_uint4(0u, 0u, 0u, 0u);

  size_t qrow0 = (size_t)batch * Nq + (size_t)qt * 16;
  size_t kv0   = (size_t)(batch / kv_div) * kv_stride;   // token base for K rows / Vt cols

  // Q^T B-fragments (persistent): runs of 16 halves at d = 32c + 16*hi
  const _Float16* qrow = Qm + (qrow0 + l15) * CDIM + head * DHEAD;
  v16h bq[3];
  for (int c = 0; c < 3; ++c) {
    int d0 = 32 * c + 16 * hi;
    H16 t;
    if (d0 < DHEAD) {
      t.u[0] = *(const uint4*)(qrow + d0);
      t.u[1] = *(const uint4*)(qrow + d0 + 8);
    } else { t.u[0] = z4; t.u[1] = z4; }
    bq[c] = t.h;
  }

  v8f ot[5];
  for (int c = 0; c < 5; ++c) ot[c] = v8f_zero();
  float m = -3.0e38f, l = 0.f;

  int ntiles = (Nkv + 31) >> 5;
  for (int kt = 0; kt < ntiles; ++kt) {
    const _Float16* krow0 = Km + (kv0 + kt * 32 + l15) * CDIM + head * DHEAD;
    const _Float16* krow1 = krow0 + 16 * CDIM;
    __builtin_prefetch(krow0 + 32 * CDIM, 0, 1);
    __builtin_prefetch(krow1 + 32 * CDIM, 0, 1);
    v8f st0 = v8f_zero(), st1 = v8f_zero();
    for (int c = 0; c < 3; ++c) {
      int o1 = 32 * c + 16 * hi;     // 16-half run d = 32c+16*hi
      H16 a0, a1;
      if (o1 < DHEAD) {              // c==2, hi==1 -> pad
        a0.u[0] = *(const uint4*)(krow0 + o1);
        a1.u[0] = *(const uint4*)(krow1 + o1);
        a0.u[1] = *(const uint4*)(krow0 + o1 + 8);
        a1.u[1] = *(const uint4*)(krow1 + o1 + 8);
      } else { a0.u[0] = z4; a1.u[0] = z4; a0.u[1] = z4; a1.u[1] = z4; }
      st0 = __builtin_amdgcn_wmma_f32_16x16x32_f16(false, a0.h, false, bq[c], (short)0, st0, false, false);
      st1 = __builtin_amdgcn_wmma_f32_16x16x32_f16(false, a1.h, false, bq[c], (short)0, st1, false, false);
    }
    float e0[8], e1[8];
    float mnew = m;
    for (int r = 0; r < 8; ++r) {
      int kr = r + 8 * hi;
      float s0 = (kt * 32 + kr      < Nkv) ? st0[r] * scale : -3.0e38f;
      float s1 = (kt * 32 + 16 + kr < Nkv) ? st1[r] * scale : -3.0e38f;
      e0[r] = s0; e1[r] = s1;
      mnew = fmaxf(mnew, fmaxf(s0, s1));
    }
    mnew = fmaxf(mnew, __shfl_xor(mnew, 16, 32));
    float alpha = __expf(m - mnew);
    float psum = 0.f;
    for (int r = 0; r < 8; ++r) {
      e0[r] = __expf(e0[r] - mnew);
      e1[r] = __expf(e1[r] - mnew);
      psum += e0[r] + e1[r];
    }
    psum += __shfl_xor(psum, 16, 32);
    l = l * alpha + psum;
    m = mnew;
    // P^T B-fragment: half-wave exchange only
    v16h pb;
    for (int r = 0; r < 8; ++r) {
      float x0 = __shfl_xor(e0[r], 16, 32);
      float x1 = __shfl_xor(e1[r], 16, 32);
      pb[r]     = (_Float16)(hi ? x1 : e0[r]);
      pb[8 + r] = (_Float16)(hi ? e1[r] : x0);
    }
    for (int c = 0; c < 5; ++c)
      for (int r = 0; r < 8; ++r) ot[c][r] *= alpha;
    // O^T += V^T P^T, V^T frags straight from transposed global V
    for (int c = 0; c < 5; ++c) {
      const _Float16* vrow = Vt + (size_t)(head * DHEAD + 16 * c + l15) * ldv + kv0 + kt * 32;
      H16 av;
      av.u[0] = *(const uint4*)(vrow + 2 * base);
      av.u[1] = *(const uint4*)(vrow + 2 * base + 8);
      ot[c] = __builtin_amdgcn_wmma_f32_16x16x32_f16(false, av.h, false, pb, (short)0, ot[c], false, false);
    }
  }
  float inv = 1.0f / l;
  _Float16* orow = Om + (qrow0 + l15) * CDIM + head * DHEAD;
  for (int c = 0; c < 5; ++c) {
    H8 h8;
    for (int r = 0; r < 8; ++r) h8.h[r] = (_Float16)(ot[c][r] * inv);
    *(uint4*)(orow + 16 * c + 8 * hi) = h8.u;   // 8 contiguous halves
  }
}

// ---------------------------------------------------------------- temporal attention (t=16, rel bias)
__global__ __launch_bounds__(32)
void temporal_attn_kernel(const _Float16* __restrict__ Qm, const _Float16* __restrict__ Km,
                          const _Float16* __restrict__ Vt, const float* __restrict__ rk,
                          const float* __restrict__ rv, _Float16* __restrict__ Om,
                          int ldv, float scale) {
  int head = blockIdx.x % NHEADS;
  int bat  = blockIdx.x / NHEADS;
  int lane = threadIdx.x;
  __shared__ float qs[16][80], ks[16][80], vs[16][80];
  __shared__ float sim[16][16];
  for (int j = lane; j < 16 * 80; j += 32) {
    int t = j / 80, d = j % 80;
    size_t idx = ((size_t)bat * TT + t) * CDIM + head * DHEAD + d;
    qs[t][d] = (float)Qm[idx];
    ks[t][d] = (float)Km[idx];
    vs[t][d] = (float)Vt[(size_t)(head * DHEAD + d) * ldv + (size_t)bat * TT + t];
  }
  __syncthreads();
  for (int p = 0; p < 8; ++p) {
    int e = lane * 8 + p;
    int t = e >> 4, s = e & 15;
    int ridx = s - t;
    ridx = ridx < -16 ? -16 : (ridx > 16 ? 16 : ridx);
    ridx += 16;
    const float* rkp = rk + ridx * DHEAD;
    float acc = 0.f;
    for (int d = 0; d < 80; ++d) acc += qs[t][d] * (ks[s][d] + rkp[d]);
    sim[t][s] = acc * scale;
  }
  __syncthreads();
  if (lane < 16) {
    int t = lane;
    float mx = -3.0e38f;
    for (int s = 0; s < 16; ++s) mx = fmaxf(mx, sim[t][s]);
    float sum = 0.f;
    for (int s = 0; s < 16; ++s) { float e = __expf(sim[t][s] - mx); sim[t][s] = e; sum += e; }
    float inv = 1.f / sum;
    for (int s = 0; s < 16; ++s) sim[t][s] *= inv;
  }
  __syncthreads();
  for (int p = 0; p < 40; ++p) {
    int e = lane * 40 + p;
    int t = e / 80, d = e % 80;
    float acc = 0.f;
    for (int s = 0; s < 16; ++s) {
      int ridx = s - t;
      ridx = ridx < -16 ? -16 : (ridx > 16 ? 16 : ridx);
      ridx += 16;
      acc += sim[t][s] * (vs[s][d] + rv[ridx * DHEAD + d]);
    }
    Om[((size_t)bat * TT + t) * CDIM + head * DHEAD + d] = (_Float16)acc;
  }
}

// ---------------------------------------------------------------- host
extern "C" void kernel_launch(void* const* d_in, const int* in_sizes, int n_in,
                              void* d_out, int out_size, void* d_ws, size_t ws_size,
                              hipStream_t stream) {
  (void)in_sizes; (void)n_in; (void)out_size; (void)ws_size;
  const float* x       = (const float*)d_in[0];
  const float* context = (const float*)d_in[1];
  const float* g1 = (const float*)d_in[2];  const float* b1 = (const float*)d_in[3];
  const float* g2 = (const float*)d_in[4];  const float* b2 = (const float*)d_in[5];
  const float* g3 = (const float*)d_in[6];  const float* b3 = (const float*)d_in[7];
  const float* g4 = (const float*)d_in[8];  const float* b4 = (const float*)d_in[9];
  const float* g5 = (const float*)d_in[10]; const float* b5 = (const float*)d_in[11];
  const float* a1_wq = (const float*)d_in[12]; const float* a1_wk = (const float*)d_in[13];
  const float* a1_wv = (const float*)d_in[14]; const float* a1_wo = (const float*)d_in[15];
  const float* a1_bo = (const float*)d_in[16];
  const float* a2_wq = (const float*)d_in[17]; const float* a2_wk = (const float*)d_in[18];
  const float* a2_wv = (const float*)d_in[19]; const float* a2_wo = (const float*)d_in[20];
  const float* a2_bo = (const float*)d_in[21];
  const float* t1_wq = (const float*)d_in[22]; const float* t1_wk = (const float*)d_in[23];
  const float* t1_wv = (const float*)d_in[24]; const float* t1_wo = (const float*)d_in[25];
  const float* t1_bo = (const float*)d_in[26];
  const float* t1_rk = (const float*)d_in[27]; const float* t1_rv = (const float*)d_in[28];
  const float* t2_wq = (const float*)d_in[29]; const float* t2_wk = (const float*)d_in[30];
  const float* t2_wv = (const float*)d_in[31]; const float* t2_wo = (const float*)d_in[32];
  const float* t2_bo = (const float*)d_in[33];
  const float* t2_rk = (const float*)d_in[34]; const float* t2_rv = (const float*)d_in[35];
  const float* ff_w1 = (const float*)d_in[36]; const float* ff_b1 = (const float*)d_in[37];
  const float* ff_w2 = (const float*)d_in[38]; const float* ff_b2 = (const float*)d_in[39];
  float* out = (float*)d_out;

  char* wsb = (char*)d_ws;
  size_t cur = 0;
  auto alloc = [&](size_t bytes) -> void* {
    void* p = wsb + cur;
    cur += (bytes + 255) & ~(size_t)255;
    return p;
  };
  const size_t W640 = (size_t)CDIM * CDIM;
  const size_t WCTX = (size_t)CTXD * CDIM;
  _Float16* w_a1q = (_Float16*)alloc(W640 * 2);
  _Float16* w_a1k = (_Float16*)alloc(W640 * 2);
  _Float16* w_a1v = (_Float16*)alloc(W640 * 2);
  _Float16* w_a1o = (_Float16*)alloc(W640 * 2);
  _Float16* w_a2q = (_Float16*)alloc(W640 * 2);
  _Float16* w_a2k = (_Float16*)alloc(WCTX * 2);
  _Float16* w_a2v = (_Float16*)alloc(WCTX * 2);
  _Float16* w_a2o = (_Float16*)alloc(W640 * 2);
  _Float16* w_t1q = (_Float16*)alloc(W640 * 2);
  _Float16* w_t1k = (_Float16*)alloc(W640 * 2);
  _Float16* w_t1v = (_Float16*)alloc(W640 * 2);
  _Float16* w_t1o = (_Float16*)alloc(W640 * 2);
  _Float16* w_t2q = (_Float16*)alloc(W640 * 2);
  _Float16* w_t2k = (_Float16*)alloc(W640 * 2);
  _Float16* w_t2v = (_Float16*)alloc(W640 * 2);
  _Float16* w_t2o = (_Float16*)alloc(W640 * 2);
  _Float16* w_ff1 = (_Float16*)alloc((size_t)CDIM * FF1N * 2);
  _Float16* w_ff2 = (_Float16*)alloc((size_t)FFIN * CDIM * 2);
  _Float16* ctx16 = (_Float16*)alloc((size_t)NB * CTXP * CTXD * 2);
  float*    xs    = (float*)alloc((size_t)R_TOK * CDIM * 4);
  float*    xt    = (float*)alloc((size_t)R_TOK * CDIM * 4);
  _Float16* ln16  = (_Float16*)alloc((size_t)R_TOK * CDIM * 2);
  _Float16* q16   = (_Float16*)alloc((size_t)R_TOK * CDIM * 2);
  _Float16* k16   = (_Float16*)alloc((size_t)R_TOK * CDIM * 2);
  _Float16* vt16  = (_Float16*)alloc((size_t)R_TOK * CDIM * 2);  // [channel][token]
  _Float16* ao16  = (_Float16*)alloc((size_t)R_TOK * CDIM * 2);
  _Float16* p16   = (_Float16*)alloc((size_t)R_TOK * FF1N * 2);
  _Float16* gg16  = (_Float16*)alloc((size_t)R_TOK * FFIN * 2);

  auto cvtT = [&](const float* src, _Float16* dst, int K, int N) {
    long long n = (long long)K * N;
    cvtT16_kernel<<<dim3((unsigned)((n + 255) / 256)), dim3(256), 0, stream>>>(src, dst, K, N);
  };
  auto gemm = [&](const _Float16* A, const _Float16* BwT, const float* bias,
                  const float* res, float* Cf, _Float16* Ch, _Float16* ChT,
                  int M, int N, int K, int ldct) {
    dim3 grid((M + GBM - 1) / GBM, N / GBN);
    gemm_f16_kernel<<<grid, dim3(256), 0, stream>>>(A, BwT, bias, res, Cf, Ch, ChT, M, N, K, ldct);
  };

  // convert weights (transposed) + context to f16
  cvtT(a1_wq, w_a1q, CDIM, CDIM); cvtT(a1_wk, w_a1k, CDIM, CDIM);
  cvtT(a1_wv, w_a1v, CDIM, CDIM); cvtT(a1_wo, w_a1o, CDIM, CDIM);
  cvtT(a2_wq, w_a2q, CDIM, CDIM); cvtT(a2_wk, w_a2k, CTXD, CDIM);
  cvtT(a2_wv, w_a2v, CTXD, CDIM); cvtT(a2_wo, w_a2o, CDIM, CDIM);
  cvtT(t1_wq, w_t1q, CDIM, CDIM); cvtT(t1_wk, w_t1k, CDIM, CDIM);
  cvtT(t1_wv, w_t1v, CDIM, CDIM); cvtT(t1_wo, w_t1o, CDIM, CDIM);
  cvtT(t2_wq, w_t2q, CDIM, CDIM); cvtT(t2_wk, w_t2k, CDIM, CDIM);
  cvtT(t2_wv, w_t2v, CDIM, CDIM); cvtT(t2_wo, w_t2o, CDIM, CDIM);
  cvtT(ff_w1, w_ff1, CDIM, FF1N);
  cvtT(ff_w2, w_ff2, FFIN, CDIM);
  {
    long long n = (long long)NB * CTXP * CTXD;
    cvt_ctx_kernel<<<dim3((unsigned)((n + 255) / 256)), dim3(256), 0, stream>>>(context, ctx16);
  }

  const long long NTOT = (long long)R_TOK * CDIM;
  const unsigned EB = (unsigned)((NTOT + 255) / 256);
  const float scale = 0.11180339887498948f; // 80^-0.5

  permute_in_kernel<<<dim3(EB), dim3(256), 0, stream>>>(x, xs);

  // ---- 1) spatial self-attention
  ln_rows_kernel<<<dim3(R_TOK), dim3(128), 0, stream>>>(xs, g1, b1, ln16);
  gemm(ln16, w_a1q, nullptr, nullptr, nullptr, q16, nullptr, R_TOK, CDIM, CDIM, 0);
  gemm(ln16, w_a1k, nullptr, nullptr, nullptr, k16, nullptr, R_TOK, CDIM, CDIM, 0);
  gemm(ln16, w_a1v, nullptr, nullptr, nullptr, nullptr, vt16, R_TOK, CDIM, CDIM, R_TOK);
  flash_attn_kernel<<<dim3(BTN * NHEADS * (HWN / 16)), dim3(32), 0, stream>>>(
      q16, k16, vt16, ao16, HWN, HWN, 1, HWN, R_TOK, scale);
  gemm(ao16, w_a1o, a1_bo, xs, xs, nullptr, nullptr, R_TOK, CDIM, CDIM, 0);

  // ---- 2) temporal attention 1
  s2t_kernel<<<dim3(EB), dim3(256), 0, stream>>>(xs, xt, 0);
  ln_rows_kernel<<<dim3(R_TOK), dim3(128), 0, stream>>>(xt, g4, b4, ln16);
  gemm(ln16, w_t1q, nullptr, nullptr, nullptr, q16, nullptr, R_TOK, CDIM, CDIM, 0);
  gemm(ln16, w_t1k, nullptr, nullptr, nullptr, k16, nullptr, R_TOK, CDIM, CDIM, 0);
  gemm(ln16, w_t1v, nullptr, nullptr, nullptr, nullptr, vt16, R_TOK, CDIM, CDIM, R_TOK);
  temporal_attn_kernel<<<dim3(BHW * NHEADS), dim3(32), 0, stream>>>(
      q16, k16, vt16, t1_rk, t1_rv, ao16, R_TOK, scale);
  gemm(ao16, w_t1o, t1_bo, xt, xt, nullptr, nullptr, R_TOK, CDIM, CDIM, 0);
  s2t_kernel<<<dim3(EB), dim3(256), 0, stream>>>(xt, xs, 1);

  // ---- 3) spatial cross-attention
  ln_rows_kernel<<<dim3(R_TOK), dim3(128), 0, stream>>>(xs, g2, b2, ln16);
  gemm(ln16,  w_a2q, nullptr, nullptr, nullptr, q16, nullptr, R_TOK, CDIM, CDIM, 0);
  gemm(ctx16, w_a2k, nullptr, nullptr, nullptr, k16, nullptr, NB * CTXP, CDIM, CTXD, 0);
  gemm(ctx16, w_a2v, nullptr, nullptr, nullptr, nullptr, vt16, NB * CTXP, CDIM, CTXD, NB * CTXP);
  flash_attn_kernel<<<dim3(BTN * NHEADS * (HWN / 16)), dim3(32), 0, stream>>>(
      q16, k16, vt16, ao16, HWN, CTXN, TT, CTXP, NB * CTXP, scale);
  gemm(ao16, w_a2o, a2_bo, xs, xs, nullptr, nullptr, R_TOK, CDIM, CDIM, 0);

  // ---- 4) temporal attention 2
  s2t_kernel<<<dim3(EB), dim3(256), 0, stream>>>(xs, xt, 0);
  ln_rows_kernel<<<dim3(R_TOK), dim3(128), 0, stream>>>(xt, g5, b5, ln16);
  gemm(ln16, w_t2q, nullptr, nullptr, nullptr, q16, nullptr, R_TOK, CDIM, CDIM, 0);
  gemm(ln16, w_t2k, nullptr, nullptr, nullptr, k16, nullptr, R_TOK, CDIM, CDIM, 0);
  gemm(ln16, w_t2v, nullptr, nullptr, nullptr, nullptr, vt16, R_TOK, CDIM, CDIM, R_TOK);
  temporal_attn_kernel<<<dim3(BHW * NHEADS), dim3(32), 0, stream>>>(
      q16, k16, vt16, t2_rk, t2_rv, ao16, R_TOK, scale);
  gemm(ao16, w_t2o, t2_bo, xt, xt, nullptr, nullptr, R_TOK, CDIM, CDIM, 0);
  s2t_kernel<<<dim3(EB), dim3(256), 0, stream>>>(xt, xs, 1);

  // ---- 5) GEGLU feed-forward
  ln_rows_kernel<<<dim3(R_TOK), dim3(128), 0, stream>>>(xs, g3, b3, ln16);
  gemm(ln16, w_ff1, ff_b1, nullptr, nullptr, p16, nullptr, R_TOK, FF1N, CDIM, 0);
  {
    long long n = (long long)R_TOK * FFIN;
    geglu_kernel<<<dim3((unsigned)((n + 255) / 256)), dim3(256), 0, stream>>>(p16, gg16);
  }
  gemm(gg16, w_ff2, ff_b2, xs, xs, nullptr, nullptr, R_TOK, CDIM, FFIN, 0);

  permute_out_kernel<<<dim3(EB), dim3(256), 0, stream>>>(xs, out);
}
